// AttentionBlock_44263932952640
// MI455X (gfx1250) — matrix operand order repaired and checked by reading, not compiled, not run on hardware
//
#include <hip/hip_runtime.h>
#include <hip/hip_bf16.h>

typedef __attribute__((ext_vector_type(16))) __bf16 v16bf;
typedef __attribute__((ext_vector_type(8)))  __bf16 v8bf;
typedef __attribute__((ext_vector_type(8)))  float  v8f;

__device__ __forceinline__ v8bf load8bf(const __bf16* p) { return *(const v8bf*)p; }
__device__ __forceinline__ v16bf load16bf(const __bf16* p) { return *(const v16bf*)p; }

// Build A-matrix operand (16x32 bf16) per-lane fragment from a row-major
// [row][K] buffer. lane<16: row=lane, K = {2*k0..2*k0+15} with k0=0;
// lane>=16: k0=8.  Caller passes rowptr already offset to its row and kchunk.
__device__ __forceinline__ v16bf makeA(const __bf16* rowptr, int k0) {
    v8bf lo = load8bf(rowptr + 2 * k0);
    v8bf hi = load8bf(rowptr + 2 * k0 + 8);
    return __builtin_shufflevector(lo, hi, 0,1,2,3,4,5,6,7,8,9,10,11,12,13,14,15);
}

__device__ __forceinline__ v8f wmma_bf16(v16bf a, v16bf b, v8f c) {
    return __builtin_amdgcn_wmma_f32_16x16x32_bf16(false, a, false, b, (short)0, c, false, false);
}

// ---------------- weight fp32 -> bf16 ----------------
__global__ void cvt_w_kernel(const float* __restrict__ wq, const float* __restrict__ wp,
                             __bf16* __restrict__ wqb, __bf16* __restrict__ wpb) {
    int i = blockIdx.x * blockDim.x + threadIdx.x;
    if (i < 768 * 256) wqb[i] = (__bf16)wq[i];
    if (i < 256 * 256) wpb[i] = (__bf16)wp[i];
}

// ---------------- GroupNorm -> xnT [b][t][c] bf16 ----------------
__global__ void groupnorm_kernel(const float* __restrict__ x,
                                 const float* __restrict__ gnw,
                                 const float* __restrict__ gnb,
                                 __bf16* __restrict__ xnT) {
    __shared__ float s1[256], s2[256];
    const int b = blockIdx.x >> 5;
    const int g = blockIdx.x & 31;
    const int tid = threadIdx.x;
    const float* xb = x + ((size_t)b * 256 + g * 8) * 4096;

    float sum = 0.f, ssq = 0.f;
    for (int cc = 0; cc < 8; ++cc)
        for (int t = tid; t < 4096; t += 256) {
            float v = xb[cc * 4096 + t];
            sum += v; ssq += v * v;
        }
    s1[tid] = sum; s2[tid] = ssq;
    __syncthreads();
    for (int s = 128; s > 0; s >>= 1) {
        if (tid < s) { s1[tid] += s1[tid + s]; s2[tid] += s2[tid + s]; }
        __syncthreads();
    }
    const float mean = s1[0] * (1.f / 32768.f);
    const float var  = s2[0] * (1.f / 32768.f) - mean * mean;
    const float rstd = rsqrtf(var + 1e-5f);

    float wv[8], bv[8];
    for (int cc = 0; cc < 8; ++cc) {
        float w = gnw[g * 8 + cc] * rstd;
        wv[cc] = w;
        bv[cc] = gnb[g * 8 + cc] - mean * w;
    }
    __bf16* xout = xnT + (size_t)b * 4096 * 256 + g * 8;
    for (int t = tid; t < 4096; t += 256) {
        v8bf o;
        for (int cc = 0; cc < 8; ++cc)
            o[cc] = (__bf16)(xb[cc * 4096 + t] * wv[cc] + bv[cc]);
        *(v8bf*)(xout + (size_t)t * 256) = o;
    }
}

// ---------------- QKV GEMM: [768,256]x[b,256,T] -> qT/kT [hb][t][ch], v [hb][ch][t] ----
__global__ void qkv_gemm_kernel(const __bf16* __restrict__ Wq,
                                const __bf16* __restrict__ xnT,
                                const float* __restrict__ qkvb,
                                __bf16* __restrict__ qT,
                                __bf16* __restrict__ kT,
                                __bf16* __restrict__ vN) {
    const int wave = threadIdx.x >> 5, lane = threadIdx.x & 31;
    const int gw = blockIdx.x * (blockDim.x >> 5) + wave;     // [0, 2*12*256)
    const int b  = gw / (12 * 256);
    const int rem = gw % (12 * 256);
    const int mb = rem >> 8;      // 12 blocks of 64 output rows
    const int nt = rem & 255;     // 256 t-tiles
    const int ln = lane & 15, half = lane >> 4;
    const int k0 = half * 8, kb = half * 16;
    const int t = nt * 16 + ln;

    v8f acc[4] = {v8f{}, v8f{}, v8f{}, v8f{}};
    const __bf16* brow  = xnT + ((size_t)b * 4096 + t) * 256;
    const __bf16* abase = Wq + (size_t)(mb * 64 + ln) * 256;

    // software-pipelined k-loop; sched_barrier pins prefetch loads above the WMMAs
    v16bf bcur = load16bf(brow + kb);
    v16bf acur[4];
    #pragma unroll
    for (int i = 0; i < 4; ++i) acur[i] = makeA(abase + (size_t)i * 16 * 256, k0);

    #pragma unroll
    for (int kc = 0; kc < 256; kc += 32) {
        const int kn = (kc + 32 < 256) ? kc + 32 : kc;
        v16bf bnext = load16bf(brow + kn + kb);
        v16bf anext[4];
        #pragma unroll
        for (int i = 0; i < 4; ++i) anext[i] = makeA(abase + (size_t)i * 16 * 256 + kn, k0);
        __builtin_amdgcn_sched_barrier(0);   // keep prefetch issued before compute
        #pragma unroll
        for (int i = 0; i < 4; ++i) acc[i] = wmma_bf16(acur[i], bcur, acc[i]);
        __builtin_amdgcn_sched_barrier(0);   // keep compute from sinking into next prefetch
        bcur = bnext;
        #pragma unroll
        for (int i = 0; i < 4; ++i) acur[i] = anext[i];
    }

    #pragma unroll
    for (int i = 0; i < 4; ++i) {
        for (int j = 0; j < 8; ++j) {
            int o = mb * 64 + i * 16 + j + 8 * half;
            float val = acc[i][j] + qkvb[o];
            int sec  = o >> 8;
            int oc   = o & 255;
            int head = oc >> 6, ch = oc & 63;
            size_t hb = (size_t)(b * 4 + head);
            if (sec == 0)      qT[(hb * 4096 + t) * 64 + ch] = (__bf16)(val * 0.125f);
            else if (sec == 1) kT[(hb * 4096 + t) * 64 + ch] = (__bf16)val;
            else               vN[(hb * 64 + ch) * 4096 + t] = (__bf16)val;
        }
    }
}

// ---------------- Flash attention: one wave per (head, 16-row t tile) ----------------
__global__ void attn_kernel(const __bf16* __restrict__ qT,
                            const __bf16* __restrict__ kT,
                            const __bf16* __restrict__ vN,
                            __bf16* __restrict__ hT) {
    __shared__ __bf16 pbuf[8][16][32];
    const int wave = threadIdx.x >> 5, lane = threadIdx.x & 31;
    const int gw = blockIdx.x * (blockDim.x >> 5) + wave;     // [0, 2048)
    const int hb = gw >> 8;                                   // b*4+head
    const int t0 = (gw & 255) * 16;
    const int ln = lane & 15, half = lane >> 4;
    const int k0 = half * 8, kb = half * 16;

    const __bf16* qbase = qT + ((size_t)hb * 4096 + (t0 + ln)) * 64;
    v16bf Aq0 = makeA(qbase, k0);        // ch 0..31
    v16bf Aq1 = makeA(qbase + 32, k0);   // ch 32..63

    v8f accO0 = {}, accO1 = {}, accO2 = {}, accO3 = {};
    float m[8], l[8];
    #pragma unroll
    for (int j = 0; j < 8; ++j) { m[j] = -1e30f; l[j] = 0.f; }

    const __bf16* kbaseRow = kT + (size_t)hb * 4096 * 64;
    const __bf16* vbase    = vN + (size_t)hb * 64 * 4096;
    __bf16 (*pb)[32] = pbuf[wave];

    auto loadBk = [&](int s, v16bf* bk) {
        const __bf16* kr0 = kbaseRow + (size_t)(s + ln) * 64 + kb;
        bk[0] = load16bf(kr0);
        bk[1] = load16bf(kr0 + 32);
        bk[2] = load16bf(kr0 + 16 * 64);
        bk[3] = load16bf(kr0 + 16 * 64 + 32);
    };
    auto loadBv = [&](int s, v16bf* bv) {
        const __bf16* vr = vbase + (size_t)ln * 4096 + s + kb;
        bv[0] = load16bf(vr);
        bv[1] = load16bf(vr + 16 * 4096);
        bv[2] = load16bf(vr + 32 * 4096);
        bv[3] = load16bf(vr + 48 * 4096);
    };

    v16bf Bk[4], Bkn[4], Bv[4], Bvn[4];
    loadBk(0, Bk);
    loadBv(0, Bv);

    #pragma unroll 2
    for (int s0 = 0; s0 < 4096; s0 += 32) {
        // prefetch next chunk (clamped; last iteration re-fetches, result unused)
        const int sn = (s0 + 32 < 4096) ? s0 + 32 : s0;
        loadBk(sn, Bkn);
        loadBv(sn, Bvn);
        __builtin_amdgcn_sched_barrier(0);   // prefetch stays above this iteration's work

        v8f z = {};
        v8f sA = wmma_bf16(Aq0, Bk[0], z);
        sA = wmma_bf16(Aq1, Bk[1], sA);
        v8f sB = wmma_bf16(Aq0, Bk[2], z);
        sB = wmma_bf16(Aq1, Bk[3], sB);

        float alpha[8];
        #pragma unroll
        for (int j = 0; j < 8; ++j) {
            float cm = fmaxf(sA[j], sB[j]);
            cm = fmaxf(cm, __shfl_xor(cm, 1, 32));
            cm = fmaxf(cm, __shfl_xor(cm, 2, 32));
            cm = fmaxf(cm, __shfl_xor(cm, 4, 32));
            cm = fmaxf(cm, __shfl_xor(cm, 8, 32));
            float mn = fmaxf(m[j], cm);
            alpha[j] = __expf(m[j] - mn);
            float p0 = __expf(sA[j] - mn);
            float p1 = __expf(sB[j] - mn);
            sA[j] = p0; sB[j] = p1;
            float rs = p0 + p1;
            rs += __shfl_xor(rs, 1, 32);
            rs += __shfl_xor(rs, 2, 32);
            rs += __shfl_xor(rs, 4, 32);
            rs += __shfl_xor(rs, 8, 32);
            l[j] = l[j] * alpha[j] + rs;
            m[j] = mn;
        }
        #pragma unroll
        for (int j = 0; j < 8; ++j) {
            accO0[j] *= alpha[j]; accO1[j] *= alpha[j];
            accO2[j] *= alpha[j]; accO3[j] *= alpha[j];
        }
        // P: D-layout (lane=s, vgpr=t-row) -> LDS -> A-layout (lane=t-row)
        #pragma unroll
        for (int j = 0; j < 8; ++j) {
            pb[j + 8 * half][ln]      = (__bf16)sA[j];
            pb[j + 8 * half][16 + ln] = (__bf16)sB[j];
        }
        asm volatile("s_wait_dscnt 0" ::: "memory");
        v16bf Ap = makeA(&pb[ln][0], k0);

        accO0 = wmma_bf16(Ap, Bv[0], accO0);
        accO1 = wmma_bf16(Ap, Bv[1], accO1);
        accO2 = wmma_bf16(Ap, Bv[2], accO2);
        accO3 = wmma_bf16(Ap, Bv[3], accO3);
        __builtin_amdgcn_sched_barrier(0);   // compute stays above next prefetch

        #pragma unroll
        for (int i = 0; i < 4; ++i) { Bk[i] = Bkn[i]; Bv[i] = Bvn[i]; }
    }

    const int b = hb >> 2, head = hb & 3;
    #pragma unroll
    for (int j = 0; j < 8; ++j) {
        float inv = 1.f / l[j];
        int t = t0 + j + 8 * half;
        __bf16* hrow = hT + ((size_t)b * 4096 + t) * 256 + head * 64 + ln;
        hrow[0]  = (__bf16)(accO0[j] * inv);
        hrow[16] = (__bf16)(accO1[j] * inv);
        hrow[32] = (__bf16)(accO2[j] * inv);
        hrow[48] = (__bf16)(accO3[j] * inv);
    }
}

// ---------------- Proj GEMM + bias + residual -> fp32 out ----------------
__global__ void proj_gemm_kernel(const __bf16* __restrict__ Wp,
                                 const __bf16* __restrict__ hT,
                                 const float* __restrict__ projb,
                                 const float* __restrict__ x,
                                 float* __restrict__ out) {
    const int wave = threadIdx.x >> 5, lane = threadIdx.x & 31;
    const int gw = blockIdx.x * (blockDim.x >> 5) + wave;     // [0, 2*4*256)
    const int b  = gw / (4 * 256);
    const int rem = gw % (4 * 256);
    const int mb = rem >> 8;      // 4 blocks of 64 rows
    const int nt = rem & 255;
    const int ln = lane & 15, half = lane >> 4;
    const int k0 = half * 8, kb = half * 16;
    const int t = nt * 16 + ln;

    v8f acc[4] = {v8f{}, v8f{}, v8f{}, v8f{}};
    const __bf16* brow  = hT + ((size_t)b * 4096 + t) * 256;
    const __bf16* abase = Wp + (size_t)(mb * 64 + ln) * 256;

    v16bf bcur = load16bf(brow + kb);
    v16bf acur[4];
    #pragma unroll
    for (int i = 0; i < 4; ++i) acur[i] = makeA(abase + (size_t)i * 16 * 256, k0);

    #pragma unroll
    for (int kc = 0; kc < 256; kc += 32) {
        const int kn = (kc + 32 < 256) ? kc + 32 : kc;
        v16bf bnext = load16bf(brow + kn + kb);
        v16bf anext[4];
        #pragma unroll
        for (int i = 0; i < 4; ++i) anext[i] = makeA(abase + (size_t)i * 16 * 256 + kn, k0);
        __builtin_amdgcn_sched_barrier(0);
        #pragma unroll
        for (int i = 0; i < 4; ++i) acc[i] = wmma_bf16(acur[i], bcur, acc[i]);
        __builtin_amdgcn_sched_barrier(0);
        bcur = bnext;
        #pragma unroll
        for (int i = 0; i < 4; ++i) acur[i] = anext[i];
    }

    #pragma unroll
    for (int i = 0; i < 4; ++i) {
        for (int j = 0; j < 8; ++j) {
            int o = mb * 64 + i * 16 + j + 8 * half;
            size_t idx = ((size_t)b * 256 + o) * 4096 + t;
            out[idx] = acc[i][j] + projb[o] + x[idx];
        }
    }
}

extern "C" void kernel_launch(void* const* d_in, const int* in_sizes, int n_in,
                              void* d_out, int out_size, void* d_ws, size_t ws_size,
                              hipStream_t stream) {
    const float* x     = (const float*)d_in[0];
    const float* gn_w  = (const float*)d_in[1];
    const float* gn_b  = (const float*)d_in[2];
    const float* qkv_w = (const float*)d_in[3];
    const float* qkv_b = (const float*)d_in[4];
    const float* prj_w = (const float*)d_in[5];
    const float* prj_b = (const float*)d_in[6];
    float* out = (float*)d_out;

    char* ws = (char*)d_ws;
    size_t off = 0;
    __bf16* wqb = (__bf16*)(ws + off); off += (size_t)768 * 256 * 2;            // 384 KB
    __bf16* wpb = (__bf16*)(ws + off); off += (size_t)256 * 256 * 2;            // 128 KB
    __bf16* xnT = (__bf16*)(ws + off); off += (size_t)2 * 4096 * 256 * 2;       // 4 MB
    __bf16* qT  = (__bf16*)(ws + off); off += (size_t)8 * 4096 * 64 * 2;        // 4 MB
    __bf16* kT  = (__bf16*)(ws + off); off += (size_t)8 * 4096 * 64 * 2;        // 4 MB
    __bf16* vN  = (__bf16*)(ws + off); off += (size_t)8 * 64 * 4096 * 2;        // 4 MB
    __bf16* hT  = (__bf16*)(ws + off); off += (size_t)2 * 4096 * 256 * 2;       // 4 MB

    cvt_w_kernel<<<(768 * 256 + 255) / 256, 256, 0, stream>>>(qkv_w, prj_w, wqb, wpb);
    groupnorm_kernel<<<64, 256, 0, stream>>>(x, gn_w, gn_b, xnT);
    qkv_gemm_kernel<<<768, 256, 0, stream>>>(wqb, xnT, qkv_b, qT, kT, vN);   // 6144 waves
    attn_kernel<<<256, 256, 0, stream>>>(qT, kT, vN, hT);                    // 2048 waves
    proj_gemm_kernel<<<256, 256, 0, stream>>>(wpb, hT, prj_b, x, out);       // 2048 waves
}
